// Net_63342177681506
// MI455X (gfx1250) — matrix operand order repaired and checked by reading, not compiled, not run on hardware
//
#include <hip/hip_runtime.h>
#include <hip/hip_bf16.h>

typedef unsigned short u16;
typedef unsigned int   u32;

typedef __attribute__((ext_vector_type(16))) __bf16 v16bf;
typedef __attribute__((ext_vector_type(8)))  float  v8f;

union bf16x16 { u16 us[16]; v16bf v; };

__device__ __forceinline__ u16 f32_to_bf16u(float f) {
    u32 u = __builtin_bit_cast(u32, f);
    u32 r = u + 0x7FFFu + ((u >> 16) & 1u);   // round-to-nearest-even
    return (u16)(r >> 16);
}
__device__ __forceinline__ float bf16u_to_f32(u16 h) {
    u32 u = ((u32)h) << 16;
    return __builtin_bit_cast(float, u);
}

// ---------------------------------------------------------------------------
// GEMM: Y[O,M](bf16) = relu( (W[O,I](f32) * X[I,M](bf16) + bias) * s + t )
// Tile: 128 (O) x 128 (M), K-step 32, 8 waves; each wave: 2x4 16x16 WMMA tiles
// ---------------------------------------------------------------------------
#define BM 128
#define BN 128
#define KB 32
#define KP 34   // LDS row pitch in u16 (even -> dword aligned rows)

__global__ __launch_bounds__(256)
void gemm_bf16_bn_relu(const float* __restrict__ W,
                       const u16*   __restrict__ X,
                       u16*         __restrict__ Y,
                       const float* __restrict__ bias,
                       const float* __restrict__ scale,
                       const float* __restrict__ shift,
                       int O, int I, int M)
{
    __shared__ u16 lw[BM * KP];   // W tile: [row o][k] bf16
    __shared__ u16 lx[BN * KP];   // X tile transposed: [col][k] bf16

    const int tid  = threadIdx.x;
    const int lane = tid & 31;
    const int wv   = tid >> 5;    // 0..7
    const int wm   = wv & 3;      // 4 row-groups of 32 rows
    const int wn   = wv >> 2;     // 2 col-groups of 64 cols

    const int obase = blockIdx.y * BM;
    const int nbase = blockIdx.x * BN;

    v8f acc[2][4];
    const v8f zero8 = {0.f,0.f,0.f,0.f,0.f,0.f,0.f,0.f};
#pragma unroll
    for (int i = 0; i < 2; ++i)
#pragma unroll
        for (int j = 0; j < 4; ++j) acc[i][j] = zero8;

    for (int kb = 0; kb < I; kb += KB) {
        // ---- W tile: 128 rows x 32 k, f32 -> bf16; 4 float4 per thread ----
#pragma unroll
        for (int it = 0; it < 4; ++it) {
            int f4  = tid + it * 256;        // 0..1023
            int row = f4 >> 3;               // 0..127
            int kc4 = (f4 & 7) << 2;         // 0,4,..,28
            int o   = obase + row;
            float4 w4 = make_float4(0.f, 0.f, 0.f, 0.f);
            if (o < O && (kb + kc4 + 4) <= I)   // all I are multiples of 4
                w4 = *reinterpret_cast<const float4*>(&W[(size_t)o * I + kb + kc4]);
            u16* dst = &lw[row * KP + kc4];
            dst[0] = f32_to_bf16u(w4.x);
            dst[1] = f32_to_bf16u(w4.y);
            dst[2] = f32_to_bf16u(w4.z);
            dst[3] = f32_to_bf16u(w4.w);
        }
        // ---- X tile: 32 k-rows x 128 cols bf16, store transposed ----
        {
            int krow = tid >> 3;             // 0..31
            int c16  = (tid & 7) << 4;       // 0,16,..,112
            u16 vals[16];
            if (kb + krow < I) {
                uint4 v = *reinterpret_cast<const uint4*>(
                    &X[(size_t)(kb + krow) * M + nbase + c16]);
                const u16* p = reinterpret_cast<const u16*>(&v);
#pragma unroll
                for (int q = 0; q < 16; ++q) vals[q] = p[q];
                if (kb + KB + krow < I)      // speculative prefetch of next tile
                    __builtin_prefetch(&X[(size_t)(kb + KB + krow) * M + nbase + c16], 0, 1);
            } else {
#pragma unroll
                for (int q = 0; q < 16; ++q) vals[q] = 0;
            }
#pragma unroll
            for (int q = 0; q < 16; ++q)
                lx[(c16 + q) * KP + krow] = vals[q];
        }
        __syncthreads();

        // ---- build fragments (contiguous k = 16*half + s) and issue WMMAs ----
        const int half = lane >> 4;
        const int mrow = lane & 15;
        v16bf afr[2];
#pragma unroll
        for (int mi = 0; mi < 2; ++mi) {
            const u16* rp = &lw[(wm * 32 + mi * 16 + mrow) * KP];
            bf16x16 u;
#pragma unroll
            for (int v = 0; v < 8; ++v) {
                int k0 = 2 * v + 16 * half;
                u32 w2 = *reinterpret_cast<const u32*>(&rp[k0]);
                u.us[2 * v]     = (u16)(w2 & 0xFFFFu);
                u.us[2 * v + 1] = (u16)(w2 >> 16);
            }
            afr[mi] = u.v;
        }
        const int ncol  = lane & 15;
        const int khalf = lane >> 4;
        v16bf bfr[4];
#pragma unroll
        for (int ni = 0; ni < 4; ++ni) {
            const u16* cp = &lx[(wn * 64 + ni * 16 + ncol) * KP + khalf * 16];
            bf16x16 u;
#pragma unroll
            for (int v = 0; v < 8; ++v) {
                u32 w2 = *reinterpret_cast<const u32*>(&cp[2 * v]);
                u.us[2 * v]     = (u16)(w2 & 0xFFFFu);
                u.us[2 * v + 1] = (u16)(w2 >> 16);
            }
            bfr[ni] = u.v;
        }
#pragma unroll
        for (int mi = 0; mi < 2; ++mi)
#pragma unroll
            for (int ni = 0; ni < 4; ++ni)
                acc[mi][ni] = __builtin_amdgcn_wmma_f32_16x16x32_bf16(
                    false, afr[mi], false, bfr[ni], (short)0, acc[mi][ni], false, false);
        __syncthreads();
    }

    // ---- epilogue: bias + folded BN + ReLU, store bf16 ----
    const int half = lane >> 4;
    const int nloc = lane & 15;
#pragma unroll
    for (int mi = 0; mi < 2; ++mi) {
#pragma unroll
        for (int r = 0; r < 8; ++r) {
            int o = obase + wm * 32 + mi * 16 + r + 8 * half;
            if (o < O) {
                float bv = bias[o], sv = scale[o], tv = shift[o];
#pragma unroll
                for (int ni = 0; ni < 4; ++ni) {
                    int col = nbase + wn * 64 + ni * 16 + nloc;
                    float h = acc[mi][ni][r] + bv;
                    float y = fmaxf(h * sv + tv, 0.0f);
                    Y[(size_t)o * M + col] = f32_to_bf16u(y);
                }
            }
        }
    }
}

// ---------------------------------------------------------------------------
// Glue kernels
// ---------------------------------------------------------------------------
__global__ void k_build_pcat(const float* __restrict__ Pa, const float* __restrict__ Pb,
                             u16* __restrict__ X)  // X [4, 8192]
{
    int col = blockIdx.x * blockDim.x + threadIdx.x;
    if (col >= 8192) return;
    int b = col >> 8, j = col & 255;
    float r0, r1, fl;
    if (j < 128) {
        r0 = Pa[(b * 128 + j) * 2 + 0] * (1.0f / 256.0f);
        r1 = Pa[(b * 128 + j) * 2 + 1] * (1.0f / 256.0f);
        fl = 0.0f;
    } else {
        int jj = j - 128;
        r0 = Pb[(b * 128 + jj) * 2 + 0] * (1.0f / 256.0f);
        r1 = Pb[(b * 128 + jj) * 2 + 1] * (1.0f / 256.0f);
        fl = 1.0f;
    }
    X[0 * 8192 + col] = f32_to_bf16u(r0);
    X[1 * 8192 + col] = f32_to_bf16u(r1);
    X[2 * 8192 + col] = f32_to_bf16u(fl);
    X[3 * 8192 + col] = f32_to_bf16u(fl);
}

__global__ void k_build_h(const u16* __restrict__ pcd, const float* __restrict__ ya,
                          const float* __restrict__ yb, const int* __restrict__ na,
                          const int* __restrict__ nb, u16* __restrict__ X) // X [2304,8192]
{
    size_t idx = (size_t)blockIdx.x * blockDim.x + threadIdx.x;
    if (idx >= 2304ull * 8192ull) return;
    int r = (int)(idx >> 13), col = (int)(idx & 8191);
    int b = col >> 8, j = col & 255;
    bool valid = (j < 128) ? (j < na[b]) : ((j - 128) < nb[b]);
    float v;
    if (r < 256) {
        v = bf16u_to_f32(pcd[(size_t)r * 8192 + col]);
    } else {
        int c = r - 256;
        v = (j < 128) ? ya[((size_t)b * 2048 + c) * 128 + j]
                      : yb[((size_t)b * 2048 + c) * 128 + (j - 128)];
    }
    X[idx] = f32_to_bf16u(valid ? v : 0.0f);
}

__global__ void k_maxpool(const u16* __restrict__ H, float* __restrict__ pcc) // H[3072,8192]
{
    int idx = blockIdx.x * blockDim.x + threadIdx.x; // 3072*32
    if (idx >= 3072 * 32) return;
    int c = idx >> 5, b = idx & 31;
    const u16* p = &H[(size_t)c * 8192 + b * 256];
    float m = -3.0e38f;
    for (int j = 0; j < 256; ++j) m = fmaxf(m, bf16u_to_f32(p[j]));
    pcc[idx] = m;   // layout [c*32 + b]
}

__global__ void k_build_e(const float* __restrict__ pcc, const u16* __restrict__ pcd,
                          u16* __restrict__ X) // X [3328, 8192]
{
    size_t idx = (size_t)blockIdx.x * blockDim.x + threadIdx.x;
    if (idx >= 3328ull * 8192ull) return;
    int r = (int)(idx >> 13), col = (int)(idx & 8191);
    int b = col >> 8;
    float v = (r < 3072) ? pcc[r * 32 + b]
                         : bf16u_to_f32(pcd[(size_t)(r - 3072) * 8192 + col]);
    X[idx] = f32_to_bf16u(v);
}

__global__ void k_peout(const u16* __restrict__ E, const float* __restrict__ W,
                        const float* __restrict__ Bb, float* __restrict__ fold)
{
    int t = blockIdx.x * blockDim.x + threadIdx.x; // 32*128
    if (t >= 32 * 128) return;
    int b = t >> 7, j = t & 127;
    int col = b * 256 + j;             // first N columns only (e[..., :N])
    float s0 = Bb[0], s1 = Bb[1];
    for (int c = 0; c < 256; ++c) {
        float x = bf16u_to_f32(E[(size_t)c * 8192 + col]);
        s0 = fmaf(W[c], x, s0);
        s1 = fmaf(W[256 + c], x, s1);
    }
    fold[t * 2 + 0] = s0;
    fold[t * 2 + 1] = s1;
}

// ---------------------------------------------------------------------------
// Sinkhorn: per batch block, 128 threads, cost recomputed on the fly
// ---------------------------------------------------------------------------
__global__ __launch_bounds__(128)
void k_sinkhorn(const float* __restrict__ f0, const float* __restrict__ f1,
                const int* __restrict__ ns, const int* __restrict__ ms,
                float* __restrict__ out)
{
    const int b = blockIdx.x;
    const int i = threadIdx.x;
    __shared__ float xs0[128], xs1[128], ys0[128], ys1[128], uu[128], vv[128];
    xs0[i] = f0[((size_t)b * 128 + i) * 2 + 0];
    xs1[i] = f0[((size_t)b * 128 + i) * 2 + 1];
    ys0[i] = f1[((size_t)b * 128 + i) * 2 + 0];
    ys1[i] = f1[((size_t)b * 128 + i) * 2 + 1];
    uu[i] = 0.f; vv[i] = 0.f;
    const int n = ns[b], m = ms[b];
    const float logmu = -logf((float)n), lognu = -logf((float)m);
    const float eps = 0.02f, inv_eps = 50.0f;
    __syncthreads();

    for (int it = 0; it < 10; ++it) {
        // u step (row i)
        {
            const bool rowv = (i < n);
            const float ui = uu[i];
            float mx = -3.0e38f;
            for (int j = 0; j < 128; ++j) {
                float d0 = xs0[i] - ys0[j], d1 = xs1[i] - ys1[j];
                float C = (rowv && j < m) ? (d0 * d0 + d1 * d1) : 1.0e9f;
                mx = fmaxf(mx, (-C + ui + vv[j]) * inv_eps);
            }
            float s = 0.f;
            for (int j = 0; j < 128; ++j) {
                float d0 = xs0[i] - ys0[j], d1 = xs1[i] - ys1[j];
                float C = (rowv && j < m) ? (d0 * d0 + d1 * d1) : 1.0e9f;
                s += expf((-C + ui + vv[j]) * inv_eps - mx);
            }
            float lse = mx + logf(s);
            float unew = rowv ? (eps * (logmu - lse) + ui) : 0.0f;
            __syncthreads();
            uu[i] = unew;
            __syncthreads();
        }
        // v step (col i)
        {
            const bool colv = (i < m);
            const float vj = vv[i];
            float mx = -3.0e38f;
            for (int r = 0; r < 128; ++r) {
                float d0 = xs0[r] - ys0[i], d1 = xs1[r] - ys1[i];
                float C = (colv && r < n) ? (d0 * d0 + d1 * d1) : 1.0e9f;
                mx = fmaxf(mx, (-C + uu[r] + vj) * inv_eps);
            }
            float s = 0.f;
            for (int r = 0; r < 128; ++r) {
                float d0 = xs0[r] - ys0[i], d1 = xs1[r] - ys1[i];
                float C = (colv && r < n) ? (d0 * d0 + d1 * d1) : 1.0e9f;
                s += expf((-C + uu[r] + vj) * inv_eps - mx);
            }
            float lse = mx + logf(s);
            float vnew = colv ? (eps * (lognu - lse) + vj) : 0.0f;
            __syncthreads();
            vv[i] = vnew;
            __syncthreads();
        }
    }
    // emit transport plan row i
    const bool rowv = (i < n);
    const float ui = uu[i];
    for (int j = 0; j < 128; ++j) {
        float d0 = xs0[i] - ys0[j], d1 = xs1[i] - ys1[j];
        bool valid = rowv && (j < m);
        float C = valid ? (d0 * d0 + d1 * d1) : 1.0e9f;
        float val = (-C + ui + vv[j]) * inv_eps;
        out[((size_t)b * 128 + i) * 128 + j] = valid ? expf(val) : 0.0f;
    }
}

// ---------------------------------------------------------------------------
// Launcher
// ---------------------------------------------------------------------------
extern "C" void kernel_launch(void* const* d_in, const int* in_sizes, int n_in,
                              void* d_out, int out_size, void* d_ws, size_t ws_size,
                              hipStream_t stream)
{
    (void)in_sizes; (void)n_in; (void)out_size; (void)ws_size;

    const float* y_src = (const float*)d_in[0];
    const float* y_tgt = (const float*)d_in[1];
    const float* P_src = (const float*)d_in[2];
    const float* P_tgt = (const float*)d_in[3];
    const int*  ns_src = (const int*)d_in[4];
    const int*  ns_tgt = (const int*)d_in[5];

    // params: pf0,pf1,pn0,pn1,pn2,pe0,pe1,pe2 each {w,b,s,t}; pe_out {w,b}
    const float* Wp[8]; const float* Bp[8]; const float* Sp[8]; const float* Tp[8];
    for (int l = 0; l < 8; ++l) {
        Wp[l] = (const float*)d_in[6 + 4 * l + 0];
        Bp[l] = (const float*)d_in[6 + 4 * l + 1];
        Sp[l] = (const float*)d_in[6 + 4 * l + 2];
        Tp[l] = (const float*)d_in[6 + 4 * l + 3];
    }
    const float* peow = (const float*)d_in[38];
    const float* peob = (const float*)d_in[39];

    const int Ov[8] = {  64,  256, 2048, 3072, 3072, 3072, 1024,  256 };
    const int Iv[8] = {   4,   64, 2304, 2048, 3072, 3328, 3072, 1024 };

    // workspace carve-up
    char* ws = (char*)d_ws;
    size_t off = 0;
    auto carve = [&](size_t bytes) {
        void* p = ws + off;
        off = (off + bytes + 255) & ~(size_t)255;
        return p;
    };
    u16*   bufA  = (u16*)  carve(3328ull * 8192ull * 2ull);
    u16*   bufB  = (u16*)  carve(3072ull * 8192ull * 2ull);
    u16*   pcd   = (u16*)  carve(256ull * 8192ull * 2ull);
    float* pcc   = (float*)carve(3072ull * 32ull * 4ull);
    float* fold0 = (float*)carve(32ull * 128ull * 2ull * 4ull);
    float* fold1 = (float*)carve(32ull * 128ull * 2ull * 4ull);

    auto gemm = [&](int l, const u16* X, u16* Y) {
        dim3 grid(8192 / BN, (Ov[l] + BM - 1) / BM);
        gemm_bf16_bn_relu<<<grid, dim3(256), 0, stream>>>(
            Wp[l], X, Y, Bp[l], Sp[l], Tp[l], Ov[l], Iv[l], 8192);
    };

    for (int dir = 0; dir < 2; ++dir) {
        const float* Pa = dir ? P_tgt : P_src;
        const float* Pb = dir ? P_src : P_tgt;
        const float* ya = dir ? y_tgt : y_src;
        const float* yb = dir ? y_src : y_tgt;
        const int*   na = dir ? ns_tgt : ns_src;
        const int*   nb = dir ? ns_src : ns_tgt;
        float* fold     = dir ? fold1 : fold0;

        k_build_pcat<<<8192 / 256, 256, 0, stream>>>(Pa, Pb, bufA);
        gemm(0, bufA, bufB);                 // pf0: 4 -> 64
        gemm(1, bufB, pcd);                  // pf1: 64 -> 256  (pcd kept)
        k_build_h<<<(2304 * 8192) / 256, 256, 0, stream>>>(pcd, ya, yb, na, nb, bufA);
        gemm(2, bufA, bufB);                 // pn0: 2304 -> 2048
        gemm(3, bufB, bufA);                 // pn1: 2048 -> 3072
        gemm(4, bufA, bufB);                 // pn2: 3072 -> 3072
        k_maxpool<<<(3072 * 32) / 256, 256, 0, stream>>>(bufB, pcc);
        k_build_e<<<(3328 * 8192) / 256, 256, 0, stream>>>(pcc, pcd, bufA);
        gemm(5, bufA, bufB);                 // pe0: 3328 -> 3072
        gemm(6, bufB, bufA);                 // pe1: 3072 -> 1024
        gemm(7, bufA, bufB);                 // pe2: 1024 -> 256
        k_peout<<<(32 * 128) / 256, 256, 0, stream>>>(bufB, peow, peob, fold);
    }

    k_sinkhorn<<<32, 128, 0, stream>>>(fold0, fold1, ns_src, ns_tgt, (float*)d_out);
}
